// CrossAttention_25074019074561
// MI455X (gfx1250) — compile-verified, matrix operand order checked by reading, not run on hardware
//
#include <hip/hip_runtime.h>

// ---------------------------------------------------------------------------
// CDNA5 / gfx1250 implementation. wave32, WMMA 16x16x32 f16 for every matmul.
// ---------------------------------------------------------------------------

typedef __attribute__((ext_vector_type(16))) _Float16 v16h;
typedef __attribute__((ext_vector_type(8)))  _Float16 v8h;
typedef __attribute__((ext_vector_type(8)))  float    v8f;

#define B_     8
#define C_     64
#define H_     256
#define W_     256
#define HW_    65536
#define CHW_   4194304          // C_*HW_
#define HEADS_ 8
#define D_     2048             // 256 * 8 (token dim for attention rows)
#define ROWS_  256
#define BH_    64               // B_*HEADS_

// Load a 16-half WMMA fragment following the CDNA5 16-bit operand layout:
// halves 0..7 hold K = base..base+7, halves 8..15 hold K = base+16..base+23
// (the caller folds the (lane>=16 ? +8 : 0) K offset into the pointer).
__device__ __forceinline__ v16h ld_frag16(const _Float16* p) {
  v8h lo = *(const v8h*)(p);
  v8h hi = *(const v8h*)(p + 16);
  v16h r;
#pragma unroll
  for (int i = 0; i < 8; ++i) { r[i] = lo[i]; r[i + 8] = hi[i]; }
  return r;
}

__device__ __forceinline__ v8f wmma_f16(v16h a, v16h b, v8f c) {
  // (neg_a, A, neg_b, B, c_mod, C, reuse_a, reuse_b)
  return __builtin_amdgcn_wmma_f32_16x16x32_f16(false, a, false, b, (short)0, c,
                                                false, false);
}

// ---------------------------------------------------------------------------
// 1) LayerNorm over channel dim, per spatial location. One thread per pixel.
// ---------------------------------------------------------------------------
__global__ void ln_kernel(const float* __restrict__ x, const float* __restrict__ lw,
                          const float* __restrict__ lb, float* __restrict__ y) {
  int p  = blockIdx.x * blockDim.x + threadIdx.x;   // [0, B_*HW_)
  int b  = p >> 16;
  int hw = p & (HW_ - 1);
  const float* xp = x + (size_t)b * CHW_ + hw;
  float v[C_];
  float s = 0.f, ss = 0.f;
#pragma unroll
  for (int c = 0; c < C_; ++c) {
    float t = xp[(size_t)c * HW_];
    v[c] = t; s += t; ss += t * t;
  }
  float mu  = s * (1.0f / C_);
  float var = ss * (1.0f / C_) - mu * mu;
  float inv = rsqrtf(var + 1e-5f);
  float* yp = y + (size_t)b * CHW_ + hw;
#pragma unroll
  for (int c = 0; c < C_; ++c)
    yp[(size_t)c * HW_] = (v[c] - mu) * inv * lw[c] + lb[c];
}

// ---------------------------------------------------------------------------
// 2) Multi-scale strip depthwise convs. Horizontal (1x7+1x11+1x21) -> o1,
//    vertical (7x1+11x1+21x1) -> o2. Combined 21-tap kernels, zero 'same' pad.
// ---------------------------------------------------------------------------
__global__ void conv_kernel(const float* __restrict__ x1,
    const float* __restrict__ w01, const float* __restrict__ b01,
    const float* __restrict__ w11, const float* __restrict__ b11,
    const float* __restrict__ w21, const float* __restrict__ b21,
    const float* __restrict__ w02, const float* __restrict__ b02,
    const float* __restrict__ w12, const float* __restrict__ b12,
    const float* __restrict__ w22, const float* __restrict__ b22,
    float* __restrict__ o1, float* __restrict__ o2) {
  size_t idx = (size_t)blockIdx.x * blockDim.x + threadIdx.x; // b,c,h,w flat
  int w = (int)(idx & 255);
  int h = (int)((idx >> 8) & 255);
  int c = (int)((idx >> 16) & 63);
  const float* rowp = x1 + (idx - (size_t)w);          // (b,c,h,0)
  const float* colp = x1 + (idx - (size_t)h * W_);     // (b,c,0,w)
  float a1 = b01[c] + b11[c] + b21[c];
  float a2 = b02[c] + b12[c] + b22[c];
#pragma unroll
  for (int t = -10; t <= 10; ++t) {
    float wh = w21[c * 21 + (t + 10)];
    float wv = w22[c * 21 + (t + 10)];
    if (t >= -5 && t <= 5) { wh += w11[c * 11 + (t + 5)]; wv += w12[c * 11 + (t + 5)]; }
    if (t >= -3 && t <= 3) { wh += w01[c * 7 + (t + 3)];  wv += w02[c * 7 + (t + 3)]; }
    int ww = w + t;
    if (ww >= 0 && ww < W_) a1 += wh * rowp[ww];
    int hh = h + t;
    if (hh >= 0 && hh < H_) a2 += wv * colp[(size_t)hh * W_];
  }
  o1[idx] = a1;
  o2[idx] = a2;
}

// ---------------------------------------------------------------------------
// 3) 1x1 projection: dst[b,o,hw] = sum_c pw[o,c]*src[b,c,hw] + pb[o].
//    WMMA GEMM: M=64 (out ch), K=64, N=128 pixels per block. 8 waves.
// ---------------------------------------------------------------------------
__global__ void proj_kernel(const float* __restrict__ src, float* __restrict__ dst,
                            const float* __restrict__ pw, const float* __restrict__ pb) {
  __shared__ __align__(16) _Float16 Wl[64 * 64];     // [o][c]
  __shared__ __align__(16) _Float16 XT[128 * 72];    // [pixel][c], padded stride
  int tid = threadIdx.x;
  int b   = blockIdx.x >> 9;
  int hw0 = (blockIdx.x & 511) << 7;
#pragma unroll
  for (int i = 0; i < 16; ++i) {
    int k = tid + i * 256;
    Wl[k] = (_Float16)pw[k];
  }
  const float* sp = src + (size_t)b * CHW_ + hw0;
  for (int k = tid; k < 8192; k += 256) {
    int c = k >> 7, p = k & 127;
    XT[p * 72 + c] = (_Float16)sp[(size_t)c * HW_ + p];
  }
  __syncthreads();
  int wave = tid >> 5, lane = tid & 31;
  int mt = wave >> 1, ntb = (wave & 1) * 4;
  int half = lane >> 4, lan = lane & 15;
  v8f acc[4];
#pragma unroll
  for (int j = 0; j < 4; ++j)
#pragma unroll
    for (int r = 0; r < 8; ++r)
      acc[j][r] = pb[mt * 16 + r + half * 8];
#pragma unroll
  for (int ks = 0; ks < 2; ++ks) {
    int ko = ks * 32 + half * 8;
    v16h a = ld_frag16(&Wl[(mt * 16 + lan) * 64 + ko]);
#pragma unroll
    for (int j = 0; j < 4; ++j) {
      v16h bf = ld_frag16(&XT[((ntb + j) * 16 + lan) * 72 + ko]);
      acc[j] = wmma_f16(a, bf, acc[j]);
    }
  }
  float* dp = dst + (size_t)b * CHW_ + hw0;
#pragma unroll
  for (int j = 0; j < 4; ++j) {
    int col = (ntb + j) * 16 + lan;
#pragma unroll
    for (int r = 0; r < 8; ++r) {
      int o = mt * 16 + r + half * 8;
      dp[(size_t)o * HW_ + col] = acc[j][r];
    }
  }
}

// ---------------------------------------------------------------------------
// 4) Build fp16 L2-normalized attention rows.
//    mode 0: q1=split_h(out2p)  mode 1: k1=split_h(out1p)
//    mode 2: q2=split_w(out1p)  mode 3: k2=split_w(out2p)
//    dst layout: [bh*256 + row][2048] contiguous fp16.
// ---------------------------------------------------------------------------
__global__ void qk_build_kernel(const float* __restrict__ o1p, const float* __restrict__ o2p,
                                _Float16* __restrict__ q1, _Float16* __restrict__ k1,
                                _Float16* __restrict__ q2, _Float16* __restrict__ k2) {
  __shared__ float red[256];
  int row = blockIdx.x;                 // bh*256 + i
  int bh = row >> 8, i = row & 255;
  int b = bh >> 3, head = bh & 7;
  int mode = blockIdx.y;
  const float* src = (mode == 0 || mode == 3) ? o2p : o1p;
  _Float16* dst = (mode == 0) ? q1 : (mode == 1) ? k1 : (mode == 2) ? q2 : k2;
  bool splitw = (mode >= 2);
  int tid = threadIdx.x;
  float tv[8];
  float ss = 0.f;
#pragma unroll
  for (int t = 0; t < 8; ++t) {
    int e = tid + t * 256;
    int cl = e & 7, sp = e >> 3;        // sp = w (split_h) or h (split_w)
    size_t cix = (size_t)(b * 64 + head * 8 + cl);
    size_t off = splitw ? (cix * HW_ + (size_t)sp * W_ + i)
                        : (cix * HW_ + (size_t)i * W_ + sp);
    float v = src[off];
    tv[t] = v;
    ss += v * v;
  }
  red[tid] = ss;
  __syncthreads();
  for (int s = 128; s > 0; s >>= 1) {
    if (tid < s) red[tid] += red[tid + s];
    __syncthreads();
  }
  float scale = 1.0f / fmaxf(sqrtf(red[0]), 1e-12f);
  _Float16* dp = dst + (size_t)row * D_;
#pragma unroll
  for (int t = 0; t < 8; ++t)
    dp[tid + t * 256] = (_Float16)(tv[t] * scale);
}

// ---------------------------------------------------------------------------
// 5) Attention: per block = (bh, 16-row tile). S=QK^T via WMMA into LDS,
//    row softmax (wave32 shfl reduction), O = P@V + Q via WMMA with V staged
//    through LDS transposed (coalesced b128 global loads, b128 ds reads).
// ---------------------------------------------------------------------------
__global__ void attn_kernel(const _Float16* __restrict__ q1, const _Float16* __restrict__ k1,
                            const _Float16* __restrict__ q2, const _Float16* __restrict__ k2,
                            float* __restrict__ o3, float* __restrict__ o4) {
  __shared__ __align__(16) float    S[16 * 256];    // 16 KB scores
  __shared__ __align__(16) _Float16 P[16 * 256];    // 8 KB probs
  __shared__ __align__(16) _Float16 VT[256 * 40];   // 20 KB V chunk, transposed+padded
  int br = blockIdx.y;
  size_t bhoff = (size_t)(blockIdx.x >> 4) * ROWS_ * D_;
  const _Float16* Q = ((br == 0) ? q1 : q2) + bhoff;
  const _Float16* K = ((br == 0) ? k1 : k2) + bhoff;
  float*          O = ((br == 0) ? o3 : o4) + bhoff;
  int i0 = (blockIdx.x & 15) * 16;
  int tid = threadIdx.x, wave = tid >> 5, lane = tid & 31;
  int half = lane >> 4, lan = lane & 15;

  // ---- S(16x256) = Qn @ Kn^T, K-dim = 2048 (64 WMMA k-steps) ----
  {
    v8f acc0 = {}, acc1 = {};
    int nt0 = wave * 2;
    const _Float16* qr  = Q + (size_t)(i0 + lan) * D_;
    const _Float16* kr0 = K + (size_t)(nt0 * 16 + lan) * D_;
    const _Float16* kr1 = K + (size_t)((nt0 + 1) * 16 + lan) * D_;
    for (int ks = 0; ks < 64; ++ks) {
      int ko = ks * 32 + half * 8;
      if (ks < 56) {                     // pull rows 8 k-steps ahead into L2/L0
        __builtin_prefetch(qr + ko + 256, 0, 0);
        __builtin_prefetch(kr0 + ko + 256, 0, 0);
        __builtin_prefetch(kr1 + ko + 256, 0, 0);
      }
      v16h a  = ld_frag16(qr + ko);
      v16h b0 = ld_frag16(kr0 + ko);
      v16h b1 = ld_frag16(kr1 + ko);
      acc0 = wmma_f16(a, b0, acc0);
      acc1 = wmma_f16(a, b1, acc1);
    }
#pragma unroll
    for (int r = 0; r < 8; ++r) {
      int m = r + half * 8;
      S[m * 256 + nt0 * 16 + lan]       = acc0[r];
      S[m * 256 + (nt0 + 1) * 16 + lan] = acc1[r];
    }
  }
  __syncthreads();

  // ---- softmax per row (each wave owns 2 rows) ----
#pragma unroll
  for (int rr = 0; rr < 2; ++rr) {
    int r = wave * 2 + rr;
    float mx = -3.0e38f;
    for (int j = lane; j < 256; j += 32) mx = fmaxf(mx, S[r * 256 + j]);
#pragma unroll
    for (int off = 16; off > 0; off >>= 1) mx = fmaxf(mx, __shfl_xor(mx, off, 32));
    float sum = 0.f;
    for (int j = lane; j < 256; j += 32) {
      float e = __expf(S[r * 256 + j] - mx);
      S[r * 256 + j] = e;
      sum += e;
    }
#pragma unroll
    for (int off = 16; off > 0; off >>= 1) sum += __shfl_xor(sum, off, 32);
    float inv = 1.0f / sum;
    for (int j = lane; j < 256; j += 32) P[r * 256 + j] = (_Float16)(S[r * 256 + j] * inv);
  }
  __syncthreads();

  // ---- O(16x2048) = P(16x256) @ V(256x2048) + Qn, V == K ----
  // d processed in 8 chunks of 256 cols; per chunk, 8 k-steps of 32 rows of V
  // staged into LDS transposed so B fragments come from ds_load_b128.
  for (int dc = 0; dc < 8; ++dc) {
    int dbase = dc * 256;
    int n0 = wave * 32;                  // this wave's 2 d-tiles within the chunk
    v8f acc0 = {}, acc1 = {};
    for (int ks = 0; ks < 8; ++ks) {
      __syncthreads();                   // previous VT fully consumed
      int j0 = ks * 32;
#pragma unroll
      for (int u = 0; u < 4; ++u) {      // coop: 32 rows x 256 cols -> VT[n][k]
        int e = tid + u * 256;
        int k = e >> 5;                  // V row within chunk (0..31)
        int g = (e & 31) * 8;            // col group (0..248)
        v8h vv = *(const v8h*)(K + (size_t)(j0 + k) * D_ + dbase + g);
#pragma unroll
        for (int q = 0; q < 8; ++q) VT[(g + q) * 40 + k] = vv[q];
      }
      __syncthreads();
      int ko = half * 8;
      v16h a  = ld_frag16(&P[lan * 256 + j0 + ko]);
      v16h b0 = ld_frag16(&VT[(n0 + lan) * 40 + ko]);
      v16h b1 = ld_frag16(&VT[(n0 + 16 + lan) * 40 + ko]);
      acc0 = wmma_f16(a, b0, acc0);
      acc1 = wmma_f16(a, b1, acc1);
    }
#pragma unroll
    for (int r = 0; r < 8; ++r) {
      int m = r + half * 8;
      size_t o0 = (size_t)(i0 + m) * D_ + dbase + n0 + lan;
      O[o0]      = acc0[r] + (float)Q[o0];
      O[o0 + 16] = acc1[r] + (float)Q[o0 + 16];
    }
  }
}

// ---------------------------------------------------------------------------
// 6) Merge heads + final double projection + residual:
//    out = W @ merge_h(o3) + W @ merge_w(o4) + 2*pb + x
// ---------------------------------------------------------------------------
__global__ void final_kernel(const float* __restrict__ o3, const float* __restrict__ o4,
                             const float* __restrict__ x, const float* __restrict__ pw,
                             const float* __restrict__ pb, float* __restrict__ out) {
  __shared__ __align__(16) _Float16 Wl[64 * 64];
  __shared__ __align__(16) _Float16 X3[128 * 72];
  __shared__ __align__(16) _Float16 X4[128 * 72];
  int tid = threadIdx.x;
  int wt = blockIdx.x & 1;
  int h  = (blockIdx.x >> 1) & 255;
  int b  = blockIdx.x >> 9;
#pragma unroll
  for (int i = 0; i < 16; ++i) {
    int k = tid + i * 256;
    Wl[k] = (_Float16)pw[k];
  }
  for (int k = tid; k < 8192; k += 256) {
    int c = k >> 7, p = k & 127;
    int w = wt * 128 + p;
    int head = c >> 3, cl = c & 7;
    size_t bhb = (size_t)(b * 8 + head) * ((size_t)ROWS_ * D_);
    X3[p * 72 + c] = (_Float16)o3[bhb + (size_t)h * D_ + w * 8 + cl];   // merge_h
    X4[p * 72 + c] = (_Float16)o4[bhb + (size_t)w * D_ + h * 8 + cl];   // merge_w
  }
  __syncthreads();
  int wave = tid >> 5, lane = tid & 31;
  int mt = wave >> 1, ntb = (wave & 1) * 4;
  int half = lane >> 4, lan = lane & 15;
  v8f acc[4];
#pragma unroll
  for (int j = 0; j < 4; ++j)
#pragma unroll
    for (int r = 0; r < 8; ++r)
      acc[j][r] = 2.0f * pb[mt * 16 + r + half * 8];
#pragma unroll
  for (int ks = 0; ks < 2; ++ks) {
    int ko = ks * 32 + half * 8;
    v16h a = ld_frag16(&Wl[(mt * 16 + lan) * 64 + ko]);
#pragma unroll
    for (int j = 0; j < 4; ++j) {
      v16h b3 = ld_frag16(&X3[((ntb + j) * 16 + lan) * 72 + ko]);
      v16h b4 = ld_frag16(&X4[((ntb + j) * 16 + lan) * 72 + ko]);
      acc[j] = wmma_f16(a, b3, acc[j]);
      acc[j] = wmma_f16(a, b4, acc[j]);
    }
  }
#pragma unroll
  for (int j = 0; j < 4; ++j) {
    int col = wt * 128 + (ntb + j) * 16 + lan;
#pragma unroll
    for (int r = 0; r < 8; ++r) {
      int o = mt * 16 + r + half * 8;
      size_t oi = (size_t)b * CHW_ + (size_t)o * HW_ + (size_t)h * W_ + col;
      out[oi] = acc[j][r] + x[oi];
    }
  }
}

// ---------------------------------------------------------------------------
// Orchestration. Workspace: 4 x 128 MiB lanes, recycled (peak 512 MiB).
//   lane A: x1 -> out2p -> out4        lane B: conv1 -> fp16 q1|k1
//   lane C: conv2 -> fp16 q2|k2        lane D: out1p -> out3
// ---------------------------------------------------------------------------
extern "C" void kernel_launch(void* const* d_in, const int* in_sizes, int n_in,
                              void* d_out, int out_size, void* d_ws, size_t ws_size,
                              hipStream_t stream) {
  (void)in_sizes; (void)n_in; (void)out_size; (void)ws_size;
  const float* x   = (const float*)d_in[0];
  const float* lw  = (const float*)d_in[1];
  const float* lb  = (const float*)d_in[2];
  const float* w01 = (const float*)d_in[3];  const float* b01 = (const float*)d_in[4];
  const float* w11 = (const float*)d_in[5];  const float* b11 = (const float*)d_in[6];
  const float* w21 = (const float*)d_in[7];  const float* b21 = (const float*)d_in[8];
  const float* w02 = (const float*)d_in[9];  const float* b02 = (const float*)d_in[10];
  const float* w12 = (const float*)d_in[11]; const float* b12 = (const float*)d_in[12];
  const float* w22 = (const float*)d_in[13]; const float* b22 = (const float*)d_in[14];
  const float* pw  = (const float*)d_in[15]; const float* pb  = (const float*)d_in[16];
  float* out = (float*)d_out;

  const size_t LANE = 134217728ull;   // 128 MiB
  char* ws = (char*)d_ws;
  float* bufA = (float*)(ws);
  float* bufB = (float*)(ws + LANE);
  float* bufC = (float*)(ws + 2 * LANE);
  float* bufD = (float*)(ws + 3 * LANE);
  _Float16* q1 = (_Float16*)bufB;
  _Float16* k1 = q1 + (size_t)BH_ * ROWS_ * D_;
  _Float16* q2 = (_Float16*)bufC;
  _Float16* k2 = q2 + (size_t)BH_ * ROWS_ * D_;
  float* o3 = bufD;
  float* o4 = bufA;

  ln_kernel<<<dim3(2048), dim3(256), 0, stream>>>(x, lw, lb, bufA);
  conv_kernel<<<dim3(131072), dim3(256), 0, stream>>>(bufA, w01, b01, w11, b11, w21, b21,
                                                      w02, b02, w12, b12, w22, b22,
                                                      bufB, bufC);
  proj_kernel<<<dim3(4096), dim3(256), 0, stream>>>(bufB, bufD, pw, pb);   // out1p
  proj_kernel<<<dim3(4096), dim3(256), 0, stream>>>(bufC, bufA, pw, pb);   // out2p
  qk_build_kernel<<<dim3(16384, 4), dim3(256), 0, stream>>>(bufD, bufA, q1, k1, q2, k2);
  attn_kernel<<<dim3(1024, 2), dim3(256), 0, stream>>>(q1, k1, q2, k2, o3, o4);
  final_kernel<<<dim3(4096), dim3(256), 0, stream>>>(o3, o4, x, pw, pb, out);
}
